// WindowAttention_23914377904603
// MI455X (gfx1250) — matrix operand order repaired and checked by reading, not compiled, not run on hardware
//
#include <hip/hip_runtime.h>
#include <hip/hip_bf16.h>

// ---------------------------------------------------------------------------
// Fused Swin window attention for gfx1250 (MI455X), bf16 WMMA path.
//   grid  = B_ (4096) workgroups, one window each
//   block = 256 threads = 8 wave32
//   LDS   = 64 KB static, phase-aliased:
//     [    0,16384)  phase1: X(bf16 64x128) -> phase2b: mask(f32 64x64)
//                    -> phase3c: P heads 0,1 (bf16 64x64 each)
//     [16384,32768)  Q(bf16 64x128, pre-scaled) -> O(bf16 64x128)
//     [32768,49152)  K(bf16 64x128) -> P heads 2,3
//     [49152,65536)  V^T(bf16 128x64)          (live until P*V done)
// ---------------------------------------------------------------------------

typedef __bf16 bf16_t;
typedef __attribute__((ext_vector_type(16))) __bf16 v16bf;
typedef __attribute__((ext_vector_type(8)))  __bf16 v8bf;
typedef __attribute__((ext_vector_type(8)))  float  v8f;

#define ATTN_SCALE 0.17677669529663687f  // 1/sqrt(32)

__device__ __forceinline__ v8f wmma_bf16(v16bf a, v16bf b, v8f c) {
  // v_wmma_f32_16x16x32_bf16  (neg_a, A, neg_b, B, c_mod, C, reuse_a, reuse_b)
  return __builtin_amdgcn_wmma_f32_16x16x32_bf16(false, a, false, b, (short)0, c,
                                                 false, false);
}

// A fragment (16x32, MxK) from row-major LDS matrix M[row][k], row stride in elems.
// Contiguous-k layout (same as B): lane L holds row rowBase+(L&15); lanes<16 -> K 0..15,
// lanes>=16 -> K 16..31 (relative to kBase).
__device__ __forceinline__ v16bf load_a_frag(const bf16_t* M, int rowBase,
                                             int stride, int kBase, int lane) {
  const int r   = rowBase + (lane & 15);
  const int off = (lane >> 4) * 16;
  const bf16_t* p = M + r * stride + kBase + off;
  v8bf lo = *(const v8bf*)(p);        // K = kBase+off   .. +7
  v8bf hi = *(const v8bf*)(p + 8);    // K = kBase+off+8 .. +15
  v16bf a;
#pragma unroll
  for (int i = 0; i < 8; ++i) { a[i] = lo[i]; a[i + 8] = hi[i]; }
  return a;
}

// B fragment (32x16, KxN) where B[k][n] = M[colBase+n][kBase+k] (contiguous in k).
// Contiguous-k layout (same as A): lane L holds column colBase+(L&15); lanes<16 -> K 0..15,
// lanes>=16 -> K 16..31 (relative to kBase), in order within the lane.
__device__ __forceinline__ v16bf load_b_frag(const bf16_t* M, int colBase,
                                             int stride, int kBase, int lane) {
  const int n  = colBase + (lane & 15);
  const int kb = kBase + (lane >> 4) * 16;
  const bf16_t* p = M + n * stride + kb;
  v8bf lo = *(const v8bf*)(p);
  v8bf hi = *(const v8bf*)(p + 8);
  v16bf b;
#pragma unroll
  for (int i = 0; i < 8; ++i) { b[i] = lo[i]; b[i + 8] = hi[i]; }
  return b;
}

// B fragment built from a global fp32 weight row (W[c][k], contiguous in k),
// converted to bf16 in-register. wrow already points at this lane's row c.
__device__ __forceinline__ v16bf load_b_frag_gw(const float* wrow, int kBase,
                                                int lane) {
  const float4* p4 = (const float4*)(wrow + kBase + ((lane >> 4) * 16));
  float4 f0 = p4[0], f1 = p4[1], f2 = p4[2], f3 = p4[3];
  v16bf b;
  b[0]  = (__bf16)f0.x; b[1]  = (__bf16)f0.y; b[2]  = (__bf16)f0.z; b[3]  = (__bf16)f0.w;
  b[4]  = (__bf16)f1.x; b[5]  = (__bf16)f1.y; b[6]  = (__bf16)f1.z; b[7]  = (__bf16)f1.w;
  b[8]  = (__bf16)f2.x; b[9]  = (__bf16)f2.y; b[10] = (__bf16)f2.z; b[11] = (__bf16)f2.w;
  b[12] = (__bf16)f3.x; b[13] = (__bf16)f3.y; b[14] = (__bf16)f3.z; b[15] = (__bf16)f3.w;
  return b;
}

// Reductions across 16-lane halves (rows of a head live in a half-wave).
__device__ __forceinline__ float half_max(float v) {
  v = fmaxf(v, __shfl_xor(v, 1, 32));
  v = fmaxf(v, __shfl_xor(v, 2, 32));
  v = fmaxf(v, __shfl_xor(v, 4, 32));
  v = fmaxf(v, __shfl_xor(v, 8, 32));
  return v;
}
__device__ __forceinline__ float half_sum(float v) {
  v += __shfl_xor(v, 1, 32);
  v += __shfl_xor(v, 2, 32);
  v += __shfl_xor(v, 4, 32);
  v += __shfl_xor(v, 8, 32);
  return v;
}

__global__ __launch_bounds__(256)
void win_attn_fused(const float* __restrict__ x,
                    const float* __restrict__ wq,   const float* __restrict__ bq,
                    const float* __restrict__ wkv,  const float* __restrict__ bkv,
                    const float* __restrict__ bias_table,
                    const float* __restrict__ wproj,const float* __restrict__ bproj,
                    const float* __restrict__ mask, int nW,
                    const int*   __restrict__ rel_index,
                    float* __restrict__ out) {
  __shared__ __align__(16) unsigned char smem[65536];
  bf16_t* sX    = (bf16_t*)(smem);
  float*  sMask = (float*)(smem);
  bf16_t* sQ    = (bf16_t*)(smem + 16384);
  bf16_t* sK    = (bf16_t*)(smem + 32768);
  bf16_t* sVt   = (bf16_t*)(smem + 49152);  // V transposed: [d=0..127][n=0..63]
  bf16_t* sO    = (bf16_t*)(smem + 16384);

  const int t    = threadIdx.x;
  const int lane = t & 31;
  const int wave = t >> 5;
  const int b    = blockIdx.x;

  // ---- Phase 1: x[b] (64x128 f32) -> LDS bf16 -------------------------------
  {
    const float4* xg = (const float4*)(x + (size_t)b * 8192);
#pragma unroll
    for (int i = 0; i < 8; ++i) {
      float4 v = xg[i * 256 + t];
      bf16_t* p = sX + (size_t)(i * 256 + t) * 4;
      p[0] = (__bf16)v.x; p[1] = (__bf16)v.y;
      p[2] = (__bf16)v.z; p[3] = (__bf16)v.w;
    }
  }
  __syncthreads();

  // ---- Phase 2: fused QKV GEMM  Y = X * [wq;wkv]^T  (64x384) ---------------
  // 24 column strips of 16; wave w handles strips w, w+8, w+16; K=128 -> 4 steps.
  for (int s = 0; s < 3; ++s) {
    const int strip = wave + s * 8;
    const int cbase = strip * 16;
    const int cl    = cbase + (lane & 15);
    const float* wrow = (cl < 128) ? (wq + (size_t)cl * 128)
                                   : (wkv + (size_t)(cl - 128) * 128);
    v8f acc[4] = {};
#pragma unroll
    for (int kk = 0; kk < 4; ++kk) {
      v16bf bf = load_b_frag_gw(wrow, kk * 32, lane);
#pragma unroll
      for (int m = 0; m < 4; ++m) {
        v16bf af = load_a_frag(sX, m * 16, 128, kk * 32, lane);
        acc[m] = wmma_bf16(af, bf, acc[m]);
      }
    }
    const float bias = (cl < 128) ? bq[cl] : bkv[cl - 128];
#pragma unroll
    for (int m = 0; m < 4; ++m) {
#pragma unroll
      for (int r = 0; r < 8; ++r) {
        const int row = m * 16 + r + (lane >> 4) * 8;
        const float v = acc[m][r] + bias;
        if (cbase < 128)      sQ[row * 128 + cl]          = (__bf16)(v * ATTN_SCALE);
        else if (cbase < 256) sK[row * 128 + (cl - 128)]  = (__bf16)v;
        else                  sVt[(cl - 256) * 64 + row]  = (__bf16)v;
      }
    }
  }
  __syncthreads();

  // ---- Phase 2b: mask[b % nW] -> LDS (reuses dead X region) ----------------
  {
    const float* mb = mask + (size_t)(b % nW) * 4096;
    for (int i = t; i < 4096; i += 256) sMask[i] = mb[i];
  }
  __syncthreads();

  // ---- Phase 3: per-wave attention. wave w: head w>>1, rows (w&1)*32..+31 --
  const int h     = wave >> 1;
  const int mbase = (wave & 1) * 32;
  const int dbase = h * 32;

  v8f S[2][4];
#pragma unroll
  for (int m2 = 0; m2 < 2; ++m2) {
    v16bf af = load_a_frag(sQ, mbase + m2 * 16, 128, dbase, lane);  // Q rows, K=d
#pragma unroll
    for (int nt = 0; nt < 4; ++nt) {
      v16bf bf = load_b_frag(sK, nt * 16, 128, dbase, lane);        // K^T cols
      v8f z = {};
      v8f sacc = wmma_bf16(af, bf, z);
#pragma unroll
      for (int r = 0; r < 8; ++r) {
        const int row = mbase + m2 * 16 + r + (lane >> 4) * 8;
        const int col = nt * 16 + (lane & 15);
        const int ri  = rel_index[row * 64 + col];       // L2-resident gather
        sacc[r] += bias_table[ri * 4 + h] + sMask[row * 64 + col];
      }
      S[m2][nt] = sacc;
    }
  }

  // Row softmax (row values live across 16-lane halves of 4 N-tiles).
#pragma unroll
  for (int m2 = 0; m2 < 2; ++m2) {
#pragma unroll
    for (int r = 0; r < 8; ++r) {
      float mx = fmaxf(fmaxf(S[m2][0][r], S[m2][1][r]),
                       fmaxf(S[m2][2][r], S[m2][3][r]));
      mx = half_max(mx);
      float sum = 0.0f;
#pragma unroll
      for (int nt = 0; nt < 4; ++nt) {
        float e = __expf(S[m2][nt][r] - mx);
        S[m2][nt][r] = e;
        sum += e;
      }
      sum = half_sum(sum);
      const float inv = 1.0f / sum;
#pragma unroll
      for (int nt = 0; nt < 4; ++nt) S[m2][nt][r] *= inv;
    }
  }

  __syncthreads();  // all waves done reading sK / sMask / sQ before aliasing

  // ---- Phase 3c: P (bf16) -> LDS, D-layout to row-major --------------------
  bf16_t* sP = (bf16_t*)(smem + ((h < 2) ? (h * 8192) : (32768 + (h - 2) * 8192)));
#pragma unroll
  for (int m2 = 0; m2 < 2; ++m2) {
#pragma unroll
    for (int nt = 0; nt < 4; ++nt) {
#pragma unroll
      for (int r = 0; r < 8; ++r) {
        const int row = mbase + m2 * 16 + r + (lane >> 4) * 8;
        const int col = nt * 16 + (lane & 15);
        sP[row * 64 + col] = (__bf16)S[m2][nt][r];
      }
    }
  }
  __syncthreads();

  // ---- Phase 3d: O = P * V  (rows 32 x d 32, K=64 -> 2 steps) --------------
  v8f O[2][2] = {};
#pragma unroll
  for (int kk = 0; kk < 2; ++kk) {
    v16bf bf0 = load_b_frag(sVt, dbase + 0,  64, kk * 32, lane);
    v16bf bf1 = load_b_frag(sVt, dbase + 16, 64, kk * 32, lane);
#pragma unroll
    for (int m2 = 0; m2 < 2; ++m2) {
      v16bf af = load_a_frag(sP, mbase + m2 * 16, 64, kk * 32, lane);
      O[m2][0] = wmma_bf16(af, bf0, O[m2][0]);
      O[m2][1] = wmma_bf16(af, bf1, O[m2][1]);
    }
  }

  // ---- Phase 4: O -> LDS bf16 (64x128, reuses dead Q region) ---------------
#pragma unroll
  for (int m2 = 0; m2 < 2; ++m2) {
#pragma unroll
    for (int dt = 0; dt < 2; ++dt) {
#pragma unroll
      for (int r = 0; r < 8; ++r) {
        const int row = mbase + m2 * 16 + r + (lane >> 4) * 8;
        const int col = dbase + dt * 16 + (lane & 15);
        sO[row * 128 + col] = (__bf16)O[m2][dt][r];
      }
    }
  }
  __syncthreads();

  // ---- Phase 5: out[b] = O * wproj^T + bproj  (64x128) ---------------------
  {
    const int cbase = wave * 16;
    const int cl    = cbase + (lane & 15);
    const float* wrow = wproj + (size_t)cl * 128;
    v8f acc[4] = {};
#pragma unroll
    for (int kk = 0; kk < 4; ++kk) {
      v16bf bf = load_b_frag_gw(wrow, kk * 32, lane);
#pragma unroll
      for (int m = 0; m < 4; ++m) {
        v16bf af = load_a_frag(sO, m * 16, 128, kk * 32, lane);
        acc[m] = wmma_bf16(af, bf, acc[m]);
      }
    }
    const float bias = bproj[cl];
    float* op = out + (size_t)b * 8192;
#pragma unroll
    for (int m = 0; m < 4; ++m) {
#pragma unroll
      for (int r = 0; r < 8; ++r) {
        const int row = m * 16 + r + (lane >> 4) * 8;
        op[row * 128 + cl] = acc[m][r] + bias;
      }
    }
  }
}

extern "C" void kernel_launch(void* const* d_in, const int* in_sizes, int n_in,
                              void* d_out, int out_size, void* d_ws, size_t ws_size,
                              hipStream_t stream) {
  (void)n_in; (void)out_size; (void)d_ws; (void)ws_size;
  const float* x          = (const float*)d_in[0];
  const float* wq         = (const float*)d_in[1];
  const float* bq         = (const float*)d_in[2];
  const float* wkv        = (const float*)d_in[3];
  const float* bkv        = (const float*)d_in[4];
  const float* bias_table = (const float*)d_in[5];
  const float* wproj      = (const float*)d_in[6];
  const float* bproj      = (const float*)d_in[7];
  const float* mask       = (const float*)d_in[8];
  const int*   rel_index  = (const int*)d_in[9];
  float* out = (float*)d_out;

  const int B_ = in_sizes[0] / (64 * 128);       // windows (4096)
  const int nW = in_sizes[8] / (64 * 64);        // mask windows (4096)

  win_attn_fused<<<B_, 256, 0, stream>>>(x, wq, bq, wkv, bkv, bias_table,
                                         wproj, bproj, mask, nW, rel_index, out);
}
